// CrossGraph_37606733644538
// MI455X (gfx1250) — compile-verified, matrix operand order checked by reading, not run on hardware
//
#include <hip/hip_runtime.h>

#define NN_ 4096
#define DD_ 1024
#define EE_ 131072
#define TAU_ 0.3f

typedef __attribute__((ext_vector_type(16))) __bf16 v16bf;
typedef __attribute__((ext_vector_type(8)))  float  v8f;
typedef int v4i_ __attribute__((vector_size(16)));   // matches async-LDS builtin param

union FragBF { uint4 q[2]; v16bf v; };

// GEMM tile config
constexpr int BM = 128, BN = 128, BK = 32, LDT = BK + 8;  // +8 bf16 pad keeps 16B align

// CDNA5 async global->LDS DMA (ASYNCcnt) if the toolchain exposes it
#if defined(__has_builtin)
#if __has_builtin(__builtin_amdgcn_global_load_async_to_lds_b128) && \
    __has_builtin(__builtin_amdgcn_s_wait_asynccnt)
#define USE_ASYNC_LDS 1
#endif
#endif

#define AS1_ __attribute__((address_space(1)))
#define AS3_ __attribute__((address_space(3)))

// ---------------- elementwise f32 -> bf16 ----------------
__global__ __launch_bounds__(256) void f32_to_bf16_k(const float* __restrict__ in,
                                                     __bf16* __restrict__ out, int n) {
  int i = blockIdx.x * 256 + threadIdx.x;
  if (i < n) out[i] = (__bf16)in[i];
}

// ---------------- fused transpose + f32 -> bf16 (square d x d) ----------------
// 32x32 tiles through LDS: coalesced on both read and write.
__global__ __launch_bounds__(256) void transpose_bf16_k(const float* __restrict__ in,
                                                        __bf16* __restrict__ out, int d) {
  __shared__ float t[32][33];
  int bx = blockIdx.x * 32, by = blockIdx.y * 32;
  int tx = threadIdx.x & 31, ty = threadIdx.x >> 5;  // 32 x 8
  #pragma unroll
  for (int i = 0; i < 32; i += 8)
    t[ty + i][tx] = in[(size_t)(by + ty + i) * d + bx + tx];
  __syncthreads();
  #pragma unroll
  for (int i = 0; i < 32; i += 8)
    out[(size_t)(bx + ty + i) * d + by + tx] = (__bf16)t[tx][ty + i];
}

// ---------------- tile staging (global -> LDS), B always row-major [Nc][K] ----
__device__ __forceinline__ void stage_tiles(const __bf16* __restrict__ Amat,
                                            const __bf16* __restrict__ Bmat,
                                            __bf16* __restrict__ sA,
                                            __bf16* __restrict__ sB,
                                            int bm0, int bn0, int k0,
                                            int K, int tid) {
  #pragma unroll
  for (int i = 0; i < 2; ++i) {
    int c = tid * 2 + i;
    int r = c >> 2, o = (c & 3) * 8;  // 8 bf16 per 16B chunk
#if defined(USE_ASYNC_LDS)
    __builtin_amdgcn_global_load_async_to_lds_b128(
        (AS1_ v4i_*)(Amat + (size_t)(bm0 + r) * K + k0 + o),
        (AS3_ v4i_*)(&sA[r * LDT + o]), 0, 0);
    __builtin_amdgcn_global_load_async_to_lds_b128(
        (AS1_ v4i_*)(Bmat + (size_t)(bn0 + r) * K + k0 + o),
        (AS3_ v4i_*)(&sB[r * LDT + o]), 0, 0);
#else
    *reinterpret_cast<uint4*>(&sA[r * LDT + o]) =
        *reinterpret_cast<const uint4*>(Amat + (size_t)(bm0 + r) * K + k0 + o);
    *reinterpret_cast<uint4*>(&sB[r * LDT + o]) =
        *reinterpret_cast<const uint4*>(Bmat + (size_t)(bn0 + r) * K + k0 + o);
#endif
  }
}

__device__ __forceinline__ void wait_stage() {
#if defined(USE_ASYNC_LDS)
  __builtin_amdgcn_s_wait_asynccnt(0);
#endif
}

// ---------------- WMMA bf16 GEMM: C[M,Nc] = alpha * A[M,K] * B[Nc,K]^T -------
// 256 threads = 8 waves (wave32), wave grid 2(M)x4(N); each wave owns a 64x32
// sub-tile = 4x2 v_wmma_f32_16x16x32_bf16 accumulators. Ping-pong LDS buffers;
// tile k+1 is fetched via async-to-LDS DMA while tile k runs on the WMMA pipes,
// completion enforced with s_wait_asynccnt AFTER compute, then one barrier.
template <bool OUT_BF16>
__global__ __launch_bounds__(256) void gemm_bt_k(const __bf16* __restrict__ Amat,
                                                 const __bf16* __restrict__ Bmat,
                                                 void* __restrict__ Cmat,
                                                 int Nc, int K, float alpha) {
  __shared__ __bf16 lsA[2][BM * LDT];
  __shared__ __bf16 lsB[2][BN * LDT];   // B^T tile: [n][k]

  const int tid  = threadIdx.x;
  const int lane = tid & 31;
  const int wave = tid >> 5;
  const int wm   = wave & 1;        // M direction, 64 rows each
  const int wn   = wave >> 1;       // N direction, 32 cols each
  const int hl   = lane >> 4;       // lane half (ISA 7.12.2 fragment layout)
  const int l15  = lane & 15;

  const int bm0 = blockIdx.x * BM;
  const int bn0 = blockIdx.y * BN;

  v8f acc[4][2] = {};

  // prologue: stage tile 0 into buffer 0
  stage_tiles(Amat, Bmat, lsA[0], lsB[0], bm0, bn0, 0, K, tid);
  wait_stage();
  __syncthreads();

  for (int k0 = 0, buf = 0; k0 < K; k0 += BK, buf ^= 1) {
    // overlap: DMA next tile into the other buffer while computing this one
    if (k0 + BK < K)
      stage_tiles(Amat, Bmat, lsA[buf ^ 1], lsB[buf ^ 1], bm0, bn0, k0 + BK, K, tid);

    // speculative prefetch two tiles ahead -> global_prefetch_b8
    if (k0 + 2 * BK < K) {
      int r = tid >> 1, o = (tid & 1) * 16;
      __builtin_prefetch(Amat + (size_t)(bm0 + r) * K + k0 + 2 * BK + o, 0, 1);
      __builtin_prefetch(Bmat + (size_t)(bn0 + r) * K + k0 + 2 * BK + o, 0, 1);
    }

    // fragments per ISA 7.12.2: lane(hl,l15); dwords 0-3 -> K = 8*hl..,
    // dwords 4-7 -> K = 16+8*hl.. : two 16B LDS reads each
    const __bf16* cA = lsA[buf];
    const __bf16* cB = lsB[buf];
    FragBF aF[4], bF[2];
    #pragma unroll
    for (int tm = 0; tm < 4; ++tm) {
      const uint4* p = reinterpret_cast<const uint4*>(&cA[(wm * 64 + tm * 16 + l15) * LDT + hl * 8]);
      aF[tm].q[0] = p[0];
      aF[tm].q[1] = p[2];
    }
    #pragma unroll
    for (int tn = 0; tn < 2; ++tn) {
      const uint4* p = reinterpret_cast<const uint4*>(&cB[(wn * 32 + tn * 16 + l15) * LDT + hl * 8]);
      bF[tn].q[0] = p[0];
      bF[tn].q[1] = p[2];
    }
    #pragma unroll
    for (int tm = 0; tm < 4; ++tm)
      #pragma unroll
      for (int tn = 0; tn < 2; ++tn)
        acc[tm][tn] = __builtin_amdgcn_wmma_f32_16x16x32_bf16(
            false, aF[tm].v, false, bF[tn].v, (short)0, acc[tm][tn], false, false);

    wait_stage();      // own next-tile DMA done (overlapped with WMMA above)
    __syncthreads();   // all waves' DMA visible; cur buffer free for reuse
  }

  // store: C/D layout -> row = v + 8*hl, col = l15
  #pragma unroll
  for (int tm = 0; tm < 4; ++tm)
    #pragma unroll
    for (int tn = 0; tn < 2; ++tn)
      #pragma unroll
      for (int v = 0; v < 8; ++v) {
        int gr = bm0 + wm * 64 + tm * 16 + v + 8 * hl;
        int gc = bn0 + wn * 32 + tn * 16 + l15;
        float val = acc[tm][tn][v] * alpha;
        if (OUT_BF16)
          ((__bf16*)Cmat)[(size_t)gr * Nc + gc] = (__bf16)val;
        else
          ((float*)Cmat)[(size_t)gr * Nc + gc] = val;
      }
}

// ---------------- Sinkhorn ----------------
// MODE 0: v = exp(s); MODE 1: v = s / colsum[c]; then row-normalize in-block.
template <int MODE>
__global__ __launch_bounds__(256) void rownorm_k(float* __restrict__ s,
                                                 const float* __restrict__ colsum, int n) {
  float* row = s + (size_t)blockIdx.x * n;
  float vals[16];
  float psum = 0.f;
  #pragma unroll
  for (int i = 0; i < 16; ++i) {
    int c = i * 256 + threadIdx.x;
    float v = row[c];
    if (MODE == 0) v = __expf(v);
    else           v = v / colsum[c];
    vals[i] = v;
    psum += v;
  }
  #pragma unroll
  for (int o = 16; o > 0; o >>= 1) psum += __shfl_down(psum, o, 32);
  __shared__ float ws[8];
  if ((threadIdx.x & 31) == 0) ws[threadIdx.x >> 5] = psum;
  __syncthreads();
  float tot = 0.f;
  #pragma unroll
  for (int w = 0; w < 8; ++w) tot += ws[w];
  float inv = 1.0f / tot;
  #pragma unroll
  for (int i = 0; i < 16; ++i) row[i * 256 + threadIdx.x] = vals[i] * inv;
}

__global__ __launch_bounds__(256) void zero_f32_k(float* p, int n) {
  int i = blockIdx.x * 256 + threadIdx.x;
  if (i < n) p[i] = 0.f;
}

__global__ __launch_bounds__(256) void fill_f32_k(float* p, float v, int n) {
  int i = blockIdx.x * 256 + threadIdx.x;
  if (i < n) p[i] = v;
}

__global__ __launch_bounds__(256) void colsum_k(const float* __restrict__ s,
                                                float* __restrict__ colsum, int n) {
  int c = blockIdx.x * 256 + threadIdx.x;
  int r0 = blockIdx.y * 256;
  float acc = 0.f;
  for (int r = r0; r < r0 + 256; ++r) acc += s[(size_t)r * n + c];
  atomicAdd(&colsum[c], acc);
}

__global__ __launch_bounds__(256) void colscale_k(float* __restrict__ s,
                                                  const float* __restrict__ colsum, int n) {
  size_t idx = (size_t)blockIdx.x * 256 + threadIdx.x;
  int c = (int)(idx & (size_t)(n - 1));
  s[idx] = s[idx] / colsum[c];
}

// ---------------- GCN conv pieces ----------------
__global__ __launch_bounds__(256) void gather_w_deg_k(const float* __restrict__ dsm,
                                                      const int* __restrict__ gIdx,
                                                      const int* __restrict__ cIdx,
                                                      float* __restrict__ w,
                                                      float* __restrict__ deg,
                                                      int E, int n) {
  int e = blockIdx.x * 256 + threadIdx.x;
  if (e >= E) return;
  int gr = gIdx[e], gc = gIdx[E + e];
  float wv = dsm[(size_t)gr * n + gc];
  w[e] = wv;
  atomicAdd(&deg[cIdx[E + e]], wv);
}

__global__ __launch_bounds__(256) void dinv_k(float* deg, int n) {
  int i = blockIdx.x * 256 + threadIdx.x;
  if (i >= n) return;
  float d = deg[i];
  deg[i] = (d > 0.f) ? rsqrtf(d) : 0.f;
}

__global__ __launch_bounds__(256) void edge_norm_k(const float* __restrict__ dinv,
                                                   const float* __restrict__ w,
                                                   const int* __restrict__ cIdx,
                                                   float* __restrict__ nrm, int E) {
  int e = blockIdx.x * 256 + threadIdx.x;
  if (e >= E) return;
  nrm[e] = dinv[cIdx[e]] * w[e] * dinv[cIdx[E + e]];
}

__global__ __launch_bounds__(256) void init_out_k(const float* __restrict__ h,
                                                  const float* __restrict__ dinv,
                                                  const float* __restrict__ bias,
                                                  float* __restrict__ out,
                                                  int n, int d) {
  size_t idx = (size_t)blockIdx.x * 256 + threadIdx.x;
  int i = (int)(idx / d), c = (int)(idx % d);
  float di = dinv[i];
  out[idx] = bias[c] + h[idx] * di * di;
}

__global__ __launch_bounds__(256) void scatter_k(const float* __restrict__ h,
                                                 const int* __restrict__ cIdx,
                                                 const float* __restrict__ nrm,
                                                 float* __restrict__ out,
                                                 int E, int d) {
  int e = blockIdx.x;
  int r = cIdx[e], c = cIdx[E + e];
  float nv = nrm[e];
  const float4* hs = reinterpret_cast<const float4*>(h + (size_t)r * d);
  float* od = out + (size_t)c * d;
  float4 v = hs[threadIdx.x];
  int o = threadIdx.x * 4;
  atomicAdd(&od[o + 0], v.x * nv);
  atomicAdd(&od[o + 1], v.y * nv);
  atomicAdd(&od[o + 2], v.z * nv);
  atomicAdd(&od[o + 3], v.w * nv);
}

__global__ __launch_bounds__(256) void copy_i32_k(const int* __restrict__ in,
                                                  int* __restrict__ out, int n) {
  int i = blockIdx.x * 256 + threadIdx.x;
  if (i < n) out[i] = in[i];
}

// ---------------- driver ----------------
extern "C" void kernel_launch(void* const* d_in, const int* in_sizes, int n_in,
                              void* d_out, int out_size, void* d_ws, size_t ws_size,
                              hipStream_t stream) {
  const int N = NN_, D = DD_, E = EE_;
  const float* x   = (const float*)d_in[0];
  const float* y   = (const float*)d_in[1];
  const int*   ei1 = (const int*)d_in[2];
  const int*   ei2 = (const int*)d_in[3];
  const float* A   = (const float*)d_in[4];
  const float* W   = (const float*)d_in[5];
  const float* b   = (const float*)d_in[6];
  float* out = (float*)d_out;

  // output offsets (floats)
  const size_t o_cxy = 0;
  const size_t o_cyx = (size_t)N * D;
  const size_t o_e1  = 2 * (size_t)N * D;
  const size_t o_e2  = o_e1 + 2 * (size_t)E;
  const size_t o_dxy = o_e2 + 2 * (size_t)E;
  const size_t o_dyx = o_dxy + (size_t)N * N;

  // workspace carve-out
  size_t off = 0;
  char* ws = (char*)d_ws;
  auto alloc = [&](size_t bytes) -> char* {
    char* p = ws + off;
    off += (bytes + 255) & ~(size_t)255;
    return p;
  };
  __bf16* x_bf  = (__bf16*)alloc((size_t)N * D * 2);
  __bf16* y_bf  = (__bf16*)alloc((size_t)N * D * 2);
  __bf16* At_bf = (__bf16*)alloc((size_t)D * D * 2);   // A^T in bf16
  __bf16* Wt_bf = (__bf16*)alloc((size_t)D * D * 2);   // W^T in bf16
  __bf16* T1b   = (__bf16*)alloc((size_t)N * D * 2);
  __bf16* T2b   = (__bf16*)alloc((size_t)N * D * 2);
  float*  h_y   = (float*)alloc((size_t)N * D * 4);
  float*  h_x   = (float*)alloc((size_t)N * D * 4);
  float*  csum  = (float*)alloc((size_t)N * 4);
  float*  deg   = (float*)alloc((size_t)N * 4);
  float*  w1    = (float*)alloc((size_t)E * 4);
  float*  w2    = (float*)alloc((size_t)E * 4);
  float*  nr1   = (float*)alloc((size_t)E * 4);
  float*  nr2   = (float*)alloc((size_t)E * 4);

  const int ND = N * D;

  // 1) casts; weight matrices transposed so every GEMM is C = A * B^T
  f32_to_bf16_k<<<ND / 256, 256, 0, stream>>>(x, x_bf, ND);
  f32_to_bf16_k<<<ND / 256, 256, 0, stream>>>(y, y_bf, ND);
  dim3 tg(D / 32, D / 32);
  transpose_bf16_k<<<tg, 256, 0, stream>>>(A, At_bf, D);
  transpose_bf16_k<<<tg, 256, 0, stream>>>(W, Wt_bf, D);

  // 2) T1 = x@A, T2 = y@A   (B^T = A  since At_bf[n][k] = A[k][n])
  dim3 gNK(N / 128, D / 128), gNN(N / 128, N / 128);
  gemm_bt_k<true><<<gNK, 256, 0, stream>>>(x_bf, At_bf, T1b, D, D, 1.0f);
  gemm_bt_k<true><<<gNK, 256, 0, stream>>>(y_bf, At_bf, T2b, D, D, 1.0f);

  // 3) affinity logits into d_out (Sinkhorn runs in place there)
  gemm_bt_k<false><<<gNN, 256, 0, stream>>>(T1b, y_bf, out + o_dxy, N, D, 1.0f / TAU_);
  gemm_bt_k<false><<<gNN, 256, 0, stream>>>(T2b, x_bf, out + o_dyx, N, D, 1.0f / TAU_);

  // 4) Sinkhorn (10 iters); col-divide fused into next row-normalize
  auto sinkhorn = [&](float* s) {
    dim3 cg(N / 256, N / 256);
    rownorm_k<0><<<N, 256, 0, stream>>>(s, csum, N);          // exp + row #1
    for (int it = 0; it < 9; ++it) {
      zero_f32_k<<<N / 256, 256, 0, stream>>>(csum, N);
      colsum_k<<<cg, 256, 0, stream>>>(s, csum, N);
      rownorm_k<1><<<N, 256, 0, stream>>>(s, csum, N);        // col #it+1 + row #it+2
    }
    zero_f32_k<<<N / 256, 256, 0, stream>>>(csum, N);
    colsum_k<<<cg, 256, 0, stream>>>(s, csum, N);
    colscale_k<<<(int)(((size_t)N * N) / 256), 256, 0, stream>>>(s, csum, N);  // col #10
  };
  sinkhorn(out + o_dxy);
  sinkhorn(out + o_dyx);

  // 5) h = features @ W
  gemm_bt_k<false><<<gNK, 256, 0, stream>>>(y_bf, Wt_bf, h_y, D, D, 1.0f);
  gemm_bt_k<false><<<gNK, 256, 0, stream>>>(x_bf, Wt_bf, h_x, D, D, 1.0f);

  // 6) GCN conv: out_conv_xy = conv(y, edges_g2, w = dsm_xy[edges_g1])
  auto gcn = [&](const float* dsm, const int* gIdx, const int* cIdx,
                 const float* h, float* wbuf, float* nbuf, float* dst) {
    fill_f32_k<<<N / 256, 256, 0, stream>>>(deg, 1.0f, N);               // self-loop
    gather_w_deg_k<<<E / 256, 256, 0, stream>>>(dsm, gIdx, cIdx, wbuf, deg, E, N);
    dinv_k<<<N / 256, 256, 0, stream>>>(deg, N);
    edge_norm_k<<<E / 256, 256, 0, stream>>>(deg, wbuf, cIdx, nbuf, E);
    init_out_k<<<ND / 256, 256, 0, stream>>>(h, deg, b, dst, N, D);
    scatter_k<<<E, D / 4, 0, stream>>>(h, cIdx, nbuf, dst, E, D);
  };
  gcn(out + o_dxy, ei1, ei2, h_y, w1, nr1, out + o_cxy);
  gcn(out + o_dyx, ei2, ei1, h_x, w2, nr2, out + o_cyx);

  // 7) pass-through edge indices
  copy_i32_k<<<2 * E / 256, 256, 0, stream>>>(ei1, (int*)(out + o_e1), 2 * E);
  copy_i32_k<<<2 * E / 256, 256, 0, stream>>>(ei2, (int*)(out + o_e2), 2 * E);

  (void)in_sizes; (void)n_in; (void)out_size; (void)ws_size;
}
